// DecoderRNN_72421738545935
// MI455X (gfx1250) — compile-verified, matrix operand order checked by reading, not run on hardware
//
#include <hip/hip_runtime.h>
#include <math.h>

typedef __bf16 bf16;
typedef __bf16 v16bf __attribute__((ext_vector_type(16)));
typedef __bf16 v8bf  __attribute__((ext_vector_type(8)));
typedef float  v8f   __attribute__((ext_vector_type(8)));

#define B_    32
#define T_    64
#define TDEC_ 48
#define H_    512
#define V_    16000
#define G3_   1536   // 3*H

// ---------------------------------------------------------------------------
// bf16 WMMA tile GEMM:  C[M x N] = A[M x K] @ W[N x K]^T (+bias[n])
// One wave32 computes a 16 x (16*NT) strip: A chunk loaded once, reused by NT
// WMMAs.  A-matrix per-lane layout (CDNA5 ISA 7.12.2, 16-bit A 16x32):
//   lane l (<16):  row m0+l, K = {k..k+7, k+16..k+23}
//   lane l+16:     row m0+l, K = {k+8..k+15, k+24..k+31}
// B loaded identically from W rows (column n0+l of B == row n0+l of W).
// ---------------------------------------------------------------------------
__device__ __forceinline__ v16bf load16(const bf16* p) {
  v8bf lo = *(const v8bf*)p;
  v8bf hi = *(const v8bf*)(p + 16);
  v16bf r;
#pragma unroll
  for (int i = 0; i < 8; ++i) { r[i] = lo[i]; r[8 + i] = hi[i]; }
  return r;
}
// f32 A with optional elementwise second operand (A + A2), converted to bf16
__device__ __forceinline__ v16bf load16s(const float* p, const float* p2) {
  v8f lo = *(const v8f*)p;
  v8f hi = *(const v8f*)(p + 16);
  if (p2) {
    lo += *(const v8f*)p2;
    hi += *(const v8f*)(p2 + 16);
  }
  v16bf r;
#pragma unroll
  for (int i = 0; i < 8; ++i) { r[i] = (bf16)lo[i]; r[8 + i] = (bf16)hi[i]; }
  return r;
}
__device__ __forceinline__ v16bf load16s(const bf16* p, const bf16* p2) {
  (void)p2;  // bf16 A path never uses a second operand
  return load16(p);
}

template <typename AT, int NT>
__device__ __forceinline__ void gemm_tile(const AT* __restrict__ A, const AT* __restrict__ A2,
                                          const bf16* __restrict__ Wm,
                                          const float* __restrict__ bias, float* __restrict__ C,
                                          int K, int lda, int ldw, int ldc, int m0, int n0) {
  const int lane = threadIdx.x;
  const int half = lane >> 4;
  const int l    = lane & 15;
  const AT* arow  = A + (size_t)(m0 + l) * lda + half * 8;
  const AT* arow2 = A2 ? A2 + (size_t)(m0 + l) * lda + half * 8 : (const AT*)nullptr;
  const bf16* wrow[NT];
#pragma unroll
  for (int j = 0; j < NT; ++j)
    wrow[j] = Wm + (size_t)(n0 + j * 16 + l) * ldw + half * 8;
  v8f acc[NT];
#pragma unroll
  for (int j = 0; j < NT; ++j) acc[j] = {0.f, 0.f, 0.f, 0.f, 0.f, 0.f, 0.f, 0.f};
#pragma unroll 2
  for (int k = 0; k < K; k += 32) {
    v16bf a = load16s(arow + k, arow2 ? arow2 + k : (const AT*)nullptr);
#pragma unroll
    for (int j = 0; j < NT; ++j) {
      v16bf b = load16(wrow[j] + k);
      acc[j] = __builtin_amdgcn_wmma_f32_16x16x32_bf16(false, a, false, b,
                                                       (short)0, acc[j], false, false);
    }
  }
  const int mrow = m0 + half * 8;
#pragma unroll
  for (int j = 0; j < NT; ++j) {
    const float bv = bias ? bias[n0 + j * 16 + l] : 0.f;
#pragma unroll
    for (int r = 0; r < 8; ++r)
      C[(size_t)(mrow + r) * ldc + n0 + j * 16 + l] = acc[j][r] + bv;
  }
}

template <typename AT, int NT>
__global__ void gemm_wmma(const AT* __restrict__ A, const AT* __restrict__ A2,
                          const bf16* __restrict__ W, const float* __restrict__ bias,
                          float* __restrict__ C, int K, int lda, int ldw, int ldc) {
  gemm_tile<AT, NT>(A, A2, W, bias, C, K, lda, ldw, ldc,
                    blockIdx.y << 4, blockIdx.x * (16 * NT));
}

// Both GRU gate GEMMs in one launch: z=0 -> gi = x@Wih^T+bih, z=1 -> gh = h@Whh^T+bhh
__global__ void gru_gates(const float* __restrict__ x, const bf16* __restrict__ Wih,
                          const float* __restrict__ bih, const float* __restrict__ h,
                          const bf16* __restrict__ Whh, const float* __restrict__ bhh,
                          float* __restrict__ gi, float* __restrict__ gh) {
  const int m0 = blockIdx.y << 4;
  const int n0 = blockIdx.x * 64;
  if (blockIdx.z == 0)
    gemm_tile<float, 4>(x, nullptr, Wih, bih, gi, H_, H_, H_, G3_, m0, n0);
  else
    gemm_tile<float, 4>(h, nullptr, Whh, bhh, gh, H_, H_, H_, G3_, m0, n0);
}

// ---------------------------------------------------------------------------
// Small elementwise / reduction kernels
// ---------------------------------------------------------------------------
__global__ void cvt_f32_bf16(const float* __restrict__ s, bf16* __restrict__ d, int n) {
  int i = blockIdx.x * blockDim.x + threadIdx.x;
  if (i < n) d[i] = (bf16)s[i];
}

__global__ void fill_zero(float* __restrict__ d, int n) {
  int i = blockIdx.x * blockDim.x + threadIdx.x;
  if (i < n) d[i] = 0.f;
}

__global__ void copy_f32(const float* __restrict__ s, float* __restrict__ d, int n) {
  int i = blockIdx.x * blockDim.x + threadIdx.x;
  if (i < n) d[i] = s[i];
}

// tanhbf[bt,h] = bf16( tanh( w1[bt,h] + q2 ) ),
// mode 0: q2 full [2048 x 512];  mode 1: q2 broadcast [32 x 512] (b = bt/64)
__global__ void tanh_add(const float* __restrict__ w1, const float* __restrict__ q2,
                         bf16* __restrict__ outp, int mode) {
  int idx = blockIdx.x * blockDim.x + threadIdx.x;   // over 2048*512
  int bt  = idx >> 9;
  int h   = idx & 511;
  float q = (mode == 0) ? q2[idx] : q2[((bt >> 6) << 9) + h];
  outp[idx] = (bf16)tanhf(w1[idx] + q);
}

// In-place softmax over the 512 hidden values of one (b,t) row of u.
__global__ void softmax_rows(float* __restrict__ u) {
  __shared__ float red[512];
  float* ur = u + (size_t)blockIdx.x * H_;
  int h = threadIdx.x;
  float v = ur[h];
  red[h] = v; __syncthreads();
  for (int s = 256; s > 0; s >>= 1) { if (h < s) red[h] = fmaxf(red[h], red[h + s]); __syncthreads(); }
  float mx = red[0]; __syncthreads();
  float e = __expf(v - mx);
  red[h] = e; __syncthreads();
  for (int s = 256; s > 0; s >>= 1) { if (h < s) red[h] += red[h + s]; __syncthreads(); }
  ur[h] = e / red[0];
}

// attns[b,h] = sum_t aw[b,t,h] * enc[b,t,h]   (no barriers, pure streaming)
__global__ void attend_reduce(const float* __restrict__ aw, const float* __restrict__ enc,
                              float* __restrict__ attns) {
  int b = blockIdx.x, h = threadIdx.x;
  float acc = 0.f;
#pragma unroll 4
  for (int t = 0; t < T_; ++t) {
    const size_t row = (size_t)(b * T_ + t) * H_;
    acc += aw[row + h] * enc[row + h];
  }
  attns[b * H_ + h] = acc;
}

// x[i,h] = br[i] + sum_{j<32} Wr[i,j]*dec[j,t] + sum_{j<32} Wr[i,32+j]*attns[j,h]
__global__ void compute_x(const float* __restrict__ Wr, const float* __restrict__ br,
                          const float* __restrict__ dec, int t,
                          const float* __restrict__ attns, float* __restrict__ x) {
  int i = blockIdx.x, h = threadIdx.x;
  float acc = br[i];
#pragma unroll 8
  for (int j = 0; j < B_; ++j) acc += Wr[i * (2 * B_) + j] * dec[j * TDEC_ + t];
#pragma unroll 8
  for (int j = 0; j < B_; ++j) acc += Wr[i * (2 * B_) + B_ + j] * attns[j * H_ + h];
  x[i * H_ + h] = acc;
}

// GRU cell combine: gates in (r, z, n) order; state updated in place
__global__ void gru_combine(const float* __restrict__ gi, const float* __restrict__ gh,
                            float* __restrict__ state) {
  int b = blockIdx.x, h = threadIdx.x;
  const float* gib = gi + (size_t)b * G3_;
  const float* ghb = gh + (size_t)b * G3_;
  float ir = gib[h], iz = gib[H_ + h], in_ = gib[2 * H_ + h];
  float hr = ghb[h], hz = ghb[H_ + h], hn  = ghb[2 * H_ + h];
  float r = 1.f / (1.f + __expf(-(ir + hr)));
  float z = 1.f / (1.f + __expf(-(iz + hz)));
  float n = tanhf(in_ + r * hn);
  float hp = state[b * H_ + h];
  state[b * H_ + h] = (1.f - z) * n + z * hp;
}

// ---------------------------------------------------------------------------
extern "C" void kernel_launch(void* const* d_in, const int* in_sizes, int n_in,
                              void* d_out, int out_size, void* d_ws, size_t ws_size,
                              hipStream_t stream) {
  (void)in_sizes; (void)n_in; (void)out_size; (void)ws_size;
  const float* enc  = (const float*)d_in[0];
  const float* dec  = (const float*)d_in[1];
  const float* W1   = (const float*)d_in[2];
  const float* W2   = (const float*)d_in[3];
  const float* vT   = (const float*)d_in[4];
  const float* Wr   = (const float*)d_in[5];
  const float* br   = (const float*)d_in[6];
  const float* Wih1 = (const float*)d_in[7];
  const float* Whh1 = (const float*)d_in[8];
  const float* bih1 = (const float*)d_in[9];
  const float* bhh1 = (const float*)d_in[10];
  const float* Wih2 = (const float*)d_in[11];
  const float* Whh2 = (const float*)d_in[12];
  const float* bih2 = (const float*)d_in[13];
  const float* bhh2 = (const float*)d_in[14];
  const float* Wout = (const float*)d_in[15];
  const float* bout = (const float*)d_in[16];
  float* out = (float*)d_out;

  // ---- workspace carve-up (all 256B aligned) ----
  char* wsp = (char*)d_ws;
  size_t off = 0;
  auto alloc = [&](size_t bytes) -> void* {
    void* p = wsp + off;
    off += (bytes + 255) & ~(size_t)255;
    return p;
  };
  bf16* encb   = (bf16*)alloc((size_t)B_ * T_ * H_ * 2);
  bf16* W1b    = (bf16*)alloc((size_t)H_ * H_ * 2);
  bf16* W2b    = (bf16*)alloc((size_t)H_ * H_ * 2);
  bf16* vTb    = (bf16*)alloc((size_t)H_ * H_ * 2);
  bf16* Wih1b  = (bf16*)alloc((size_t)G3_ * H_ * 2);
  bf16* Whh1b  = (bf16*)alloc((size_t)G3_ * H_ * 2);
  bf16* Wih2b  = (bf16*)alloc((size_t)G3_ * H_ * 2);
  bf16* Whh2b  = (bf16*)alloc((size_t)G3_ * H_ * 2);
  bf16* Woutb  = (bf16*)alloc((size_t)V_ * H_ * 2);
  float* w1buf = (float*)alloc((size_t)B_ * T_ * H_ * 4);
  float* ubuf  = (float*)alloc((size_t)B_ * T_ * H_ * 4);   // q2full, then u, then aw
  bf16* tanhbf = (bf16*)alloc((size_t)B_ * T_ * H_ * 2);
  float* q2    = (float*)alloc((size_t)B_ * H_ * 4);
  float* attns = (float*)alloc((size_t)B_ * H_ * 4);
  float* xbuf  = (float*)alloc((size_t)B_ * H_ * 4);
  float* state = (float*)alloc((size_t)B_ * H_ * 4);
  float* gi    = (float*)alloc((size_t)B_ * G3_ * 4);
  float* gh    = (float*)alloc((size_t)B_ * G3_ * 4);

  auto cvt = [&](const float* s, bf16* d, int n) {
    cvt_f32_bf16<<<(n + 255) / 256, 256, 0, stream>>>(s, d, n);
  };
  // GEMM, bf16 A (K = 512 fixed, 16x64 strip per wave)
  auto gemmB = [&](const bf16* A, const bf16* Wm, const float* bias, float* C,
                   int M, int N, int ldc) {
    gemm_wmma<bf16, 4><<<dim3(N / 64, M / 16), 32, 0, stream>>>(
        A, nullptr, Wm, bias, C, H_, H_, H_, ldc);
  };
  // GEMM, f32 A (optionally A+A2) converted on the fly
  auto gemmF = [&](const float* A, const float* A2, const bf16* Wm, const float* bias,
                   float* C, int M, int N, int ldc) {
    gemm_wmma<float, 4><<<dim3(N / 64, M / 16), 32, 0, stream>>>(
        A, A2, Wm, bias, C, H_, H_, H_, ldc);
  };
  // attend: tanh(w1+q2) -> u = tanhX @ vT^T -> row softmax (in place) -> reduce over t
  auto attend = [&](int mode, const float* q2ptr) {
    tanh_add<<<(B_ * T_ * H_) / 256, 256, 0, stream>>>(w1buf, q2ptr, tanhbf, mode);
    gemmB(tanhbf, vTb, nullptr, ubuf, B_ * T_, H_, H_);
    softmax_rows<<<B_ * T_, H_, 0, stream>>>(ubuf);
    attend_reduce<<<B_, H_, 0, stream>>>(ubuf, enc, attns);
  };

  // ---- one-time setup ----
  cvt(enc, encb, B_ * T_ * H_);
  cvt(W1, W1b, H_ * H_);
  cvt(W2, W2b, H_ * H_);
  cvt(vT, vTb, H_ * H_);
  cvt(Wih1, Wih1b, G3_ * H_);
  cvt(Whh1, Whh1b, G3_ * H_);
  cvt(Wih2, Wih2b, G3_ * H_);
  cvt(Whh2, Whh2b, G3_ * H_);
  cvt(Wout, Woutb, V_ * H_);
  fill_zero<<<(B_ * H_ + 255) / 256, 256, 0, stream>>>(state, B_ * H_);

  gemmB(encb, W1b, nullptr, w1buf, B_ * T_, H_, H_);   // w1 (hoisted)
  gemmB(encb, W2b, nullptr, ubuf,  B_ * T_, H_, H_);   // q2full for attns0
  attend(0, ubuf);                                     // attns0

  // ---- 48 sequential decode steps ----
  for (int t = 0; t < TDEC_; ++t) {
    compute_x<<<B_, H_, 0, stream>>>(Wr, br, dec, t, attns, xbuf);

    gru_gates<<<dim3(G3_ / 64, B_ / 16, 2), 32, 0, stream>>>(
        xbuf, Wih1b, bih1, state, Whh1b, bhh1, gi, gh);
    gru_combine<<<B_, H_, 0, stream>>>(gi, gh, state);

    gru_gates<<<dim3(G3_ / 64, B_ / 16, 2), 32, 0, stream>>>(
        xbuf, Wih2b, bih2, state, Whh2b, bhh2, gi, gh);
    gru_combine<<<B_, H_, 0, stream>>>(gi, gh, state);

    gemmF(state, nullptr, W2b, nullptr, q2, B_, H_, H_);
    attend(1, q2);

    // out = (state + attns) @ Wout^T + bout   (add fused into A load)
    gemmF(state, attns, Woutb, bout, out + (size_t)t * B_ * V_, B_, V_, V_);
  }

  // final state appended after outputs
  copy_f32<<<(B_ * H_ + 255) / 256, 256, 0, stream>>>(
      state, out + (size_t)TDEC_ * B_ * V_, B_ * H_);
}
